// BinaryDecorator_52828097741445
// MI455X (gfx1250) — 8-bit matrix operand order repaired and checked by reading, compiled, not run on hardware
//
#include <hip/hip_runtime.h>


typedef __attribute__((ext_vector_type(16))) int   v16i;
typedef __attribute__((ext_vector_type(8)))  float v8f;
typedef __attribute__((ext_vector_type(4)))  int   v4i;

typedef __attribute__((address_space(1))) v4i* gas_v4i_ptr;
typedef __attribute__((address_space(3))) v4i* las_v4i_ptr;

#define NROWS 8192
#define KDIM  4096
#define ODIM  4096

#define FP8_P1 0x38
#define FP8_M1 0xB8

#define BM 256
#define BN 128
#define BK 128
#define KSTEPS (KDIM / BK)
#define LDS_STRIDE 144

#if defined(__has_builtin)
#if __has_builtin(__builtin_amdgcn_global_load_async_to_lds_b128)
#define HAVE_ASYNC_LDS 1
#endif
#endif
#ifndef HAVE_ASYNC_LDS
#define HAVE_ASYNC_LDS 0
#warning "global_load_async_to_lds_b128 builtin not available; using sync LDS staging fallback"
#endif

__global__ void __launch_bounds__(256)
binarize_x_kernel(const float* __restrict__ x,
                  unsigned char* __restrict__ x8,
                  float* __restrict__ abs_sum) {
  const int tid = blockIdx.x * 256 + threadIdx.x;
  const float4 v = ((const float4*)x)[tid];

  uchar4 o;
  o.x = v.x > 0.f ? FP8_P1 : (v.x < 0.f ? FP8_M1 : 0);
  o.y = v.y > 0.f ? FP8_P1 : (v.y < 0.f ? FP8_M1 : 0);
  o.z = v.z > 0.f ? FP8_P1 : (v.z < 0.f ? FP8_M1 : 0);
  o.w = v.w > 0.f ? FP8_P1 : (v.w < 0.f ? FP8_M1 : 0);
  ((uchar4*)x8)[tid] = o;

  float s = fabsf(v.x) + fabsf(v.y) + fabsf(v.z) + fabsf(v.w);

  __shared__ float red[256];
  red[threadIdx.x] = s;
  __syncthreads();
  #pragma unroll
  for (int off = 128; off > 0; off >>= 1) {
    if (threadIdx.x < off) red[threadIdx.x] += red[threadIdx.x + off];
    __syncthreads();
  }
  if (threadIdx.x == 0) atomicAdd(abs_sum, red[0]);
}

__global__ void __launch_bounds__(256)
binarize_w_kernel(const float* __restrict__ W,
                  unsigned char* __restrict__ w8) {
  const int tid = blockIdx.x * 256 + threadIdx.x;
  const float4 v = ((const float4*)W)[tid];
  uchar4 o;
  o.x = v.x > 0.f ? FP8_P1 : (v.x < 0.f ? FP8_M1 : 0);
  o.y = v.y > 0.f ? FP8_P1 : (v.y < 0.f ? FP8_M1 : 0);
  o.z = v.z > 0.f ? FP8_P1 : (v.z < 0.f ? FP8_M1 : 0);
  o.w = v.w > 0.f ? FP8_P1 : (v.w < 0.f ? FP8_M1 : 0);
  ((uchar4*)w8)[tid] = o;
}

__global__ void __launch_bounds__(256)
bgemm_fp8_wmma_kernel(const unsigned char* __restrict__ x8,
                      const unsigned char* __restrict__ w8,
                      const float* __restrict__ bias,
                      const float* __restrict__ abs_sum,
                      float* __restrict__ out) {
  __shared__ __align__(16) unsigned char lsB[2][BN * LDS_STRIDE];

  const int tid  = threadIdx.x;
  const int lane = tid & 31;
  const int wid  = tid >> 5;
  const int ntb  = (blockIdx.x & 31) * BN;
  const int mtb  = (blockIdx.x >> 5) * BM;

  const int n  = lane & 15;
  const int hi = lane >> 4;

  const unsigned char* aptr0 =
      x8 + (size_t)(mtb + wid * 32 + n) * KDIM + hi * 16;
  const unsigned char* aptr1 = aptr0 + (size_t)16 * KDIM;
  const unsigned char* wbase = w8 + (size_t)ntb * KDIM;

  const int jr0 = tid >> 3;
  const int jc  = (tid & 7) * 16;

#define STAGE(BUF, KB)                                                        \
  {                                                                           \
    _Pragma("unroll")                                                         \
    for (int p = 0; p < 4; ++p) {                                             \
      const int r = jr0 + p * 32;                                             \
      const unsigned char* g = wbase + (size_t)r * KDIM + (KB) + jc;          \
      unsigned char* l = &lsB[BUF][r * LDS_STRIDE + jc];                      \
      STAGE_ONE(g, l)                                                         \
    }                                                                         \
  }

#if HAVE_ASYNC_LDS
#define STAGE_ONE(G, L)                                                       \
      __builtin_amdgcn_global_load_async_to_lds_b128(                         \
          (gas_v4i_ptr)(G), (las_v4i_ptr)(L), 0, 0);
#define WAIT_ASYNC_PIPE() asm volatile("s_wait_asynccnt 0x4" ::: "memory");
#define WAIT_ASYNC_ALL()  asm volatile("s_wait_asynccnt 0x0" ::: "memory");
#else
#define STAGE_ONE(G, L)                                                       \
      { int4 q = *(const int4*)(G); *(int4*)(L) = q; }
#define WAIT_ASYNC_PIPE()
#define WAIT_ASYNC_ALL()
#endif

  v8f acc0[8], acc1[8];
  #pragma unroll
  for (int j = 0; j < 8; ++j) { acc0[j] = (v8f){}; acc1[j] = (v8f){}; }

  STAGE(0, 0)

  #pragma unroll 1
  for (int ks = 0; ks < KSTEPS; ++ks) {
    const int cur = ks & 1;
    const int kb  = ks * BK;

    if (ks + 1 < KSTEPS) {
      STAGE(1 - cur, kb + BK)
      WAIT_ASYNC_PIPE()
    } else {
      WAIT_ASYNC_ALL()
    }
    __syncthreads();

    v16i a0, a1;
    #pragma unroll
    for (int t = 0; t < 8; ++t) {
      int2 q0 = *(const int2*)(aptr0 + kb + (t >> 1) * 32 + (t & 1) * 8);
      int2 q1 = *(const int2*)(aptr1 + kb + (t >> 1) * 32 + (t & 1) * 8);
      a0[2 * t + 0] = q0.x; a0[2 * t + 1] = q0.y;
      a1[2 * t + 0] = q1.x; a1[2 * t + 1] = q1.y;
    }

    #pragma unroll
    for (int j = 0; j < 8; ++j) {
      const unsigned char* lb = &lsB[cur][(j * 16 + n) * LDS_STRIDE + hi * 16];
      v16i bf;
      #pragma unroll
      for (int t = 0; t < 4; ++t) {
        int4 q = *(const int4*)(lb + t * 32);
        bf[4 * t + 0] = q.x; bf[4 * t + 1] = q.y;
        bf[4 * t + 2] = q.z; bf[4 * t + 3] = q.w;
      }
      acc0[j] = __builtin_amdgcn_wmma_f32_16x16x128_fp8_fp8(a0, bf, (short)0,
                                                            acc0[j], false, false);
      acc1[j] = __builtin_amdgcn_wmma_f32_16x16x128_fp8_fp8(a1, bf, (short)0,
                                                            acc1[j], false, false);
    }
    __syncthreads();
  }
#undef STAGE
#undef STAGE_ONE
#undef WAIT_ASYNC_PIPE
#undef WAIT_ASYNC_ALL

  const float scale = abs_sum[0] * (1.0f / (float)((size_t)NROWS * (size_t)KDIM));
  const int row0 = mtb + wid * 32 + hi * 8;

  #pragma unroll
  for (int j = 0; j < 8; ++j) {
    const int col = ntb + j * 16 + n;
    const float bv = bias[col];
    #pragma unroll
    for (int r = 0; r < 8; ++r) {
      out[(size_t)(row0 + r) * ODIM + col]      = (acc0[j][r] + bv) * scale;
      out[(size_t)(row0 + 16 + r) * ODIM + col] = (acc1[j][r] + bv) * scale;
    }
  }
}

extern "C" void kernel_launch(void* const* d_in, const int* in_sizes, int n_in,
                              void* d_out, int out_size, void* d_ws, size_t ws_size,
                              hipStream_t stream) {
  (void)in_sizes; (void)n_in; (void)out_size; (void)ws_size;

  const float* x = (const float*)d_in[0];
  const float* W = (const float*)d_in[1];
  const float* b = (const float*)d_in[2];
  float* out = (float*)d_out;

  unsigned char* ws = (unsigned char*)d_ws;
  float* abs_sum    = (float*)ws;
  unsigned char* x8 = ws + 256;
  unsigned char* w8 = ws + 256 + (size_t)NROWS * KDIM;

  (void)hipMemsetAsync(abs_sum, 0, sizeof(float), stream);

  binarize_x_kernel<<<(NROWS * KDIM / 4) / 256, 256, 0, stream>>>(x, x8, abs_sum);
  binarize_w_kernel<<<(ODIM * KDIM / 4) / 256, 256, 0, stream>>>(W, w8);

  bgemm_fp8_wmma_kernel<<<1024, 256, 0, stream>>>(x8, w8, b, abs_sum, out);
}
